// HGTEncoder_2954937500396
// MI455X (gfx1250) — matrix operand order repaired and checked by reading, not compiled, not run on hardware
//
#include <hip/hip_runtime.h>
#include <hip/hip_bf16.h>
#include <math.h>

// ---------------------------------------------------------------------------
// Types for CDNA5 WMMA (wave32): v_wmma_f32_16x16x32_bf16
// ---------------------------------------------------------------------------
typedef __attribute__((ext_vector_type(16))) __bf16 bf16x16;
typedef __attribute__((ext_vector_type(8)))  float  f32x8;
typedef __attribute__((ext_vector_type(4)))  unsigned int u32x4;
typedef __attribute__((ext_vector_type(8)))  int    i32x8;
typedef __attribute__((ext_vector_type(4)))  int    i32x4;

union FragA { bf16x16 v; unsigned int u[8]; };

__device__ __forceinline__ unsigned short f2bf(float f) {
    unsigned int u = __builtin_bit_cast(unsigned int, f);
    unsigned int r = u + 0x7FFFu + ((u >> 16) & 1u);   // round-to-nearest-even
    return (unsigned short)(r >> 16);
}

__device__ __forceinline__ float gelu_exact(float x) {
    return 0.5f * x * (1.0f + erff(x * 0.70710678118654752f));
}

// sortable-uint encoding of float for atomicMax-based segment max
__device__ __forceinline__ unsigned int fkey(float f) {
    unsigned int b = __builtin_bit_cast(unsigned int, f);
    return (b & 0x80000000u) ? ~b : (b | 0x80000000u);
}
__device__ __forceinline__ float funkey(unsigned int k) {
    unsigned int b = (k & 0x80000000u) ? (k & 0x7FFFFFFFu) : ~k;
    return __builtin_bit_cast(float, b);
}

// ---------------------------------------------------------------------------
// Generic strided WMMA GEMM:
//   C[m, ccol+n] = act( A[m, acol+k] * W[k, n] + bias[n] )
// Block = 256 threads = 8 waves; block tile 64(M) x 64(N); K stepped by 32.
// A tile (64x32 fp32) staged to LDS either by
//   * TDM: one tensor_load_to_lds per block (wave 0 builds D#, TENSORcnt), or
//   * per-lane global_load_async_to_lds_b128 (ASYNCcnt),
// then converted to bf16 (+ optional GELU) at fragment build.
// W tile staged transposed+bf16 so lanes read packed pairs as dwords.
// Each wave holds two 16x16 f32 accumulators sharing one A fragment.
// ---------------------------------------------------------------------------
__global__ __launch_bounds__(256)
void wmma_gemm_k(const float* __restrict__ A, int lda, int acol,
                 const float* __restrict__ W, int ldw,
                 const float* __restrict__ bias,
                 float* __restrict__ C, int ldc, int ccol,
                 int M, int N, int K, int gelu_in, int gelu_out, int use_tdm) {
    __shared__ float          As[64 * 32];   // [row][k] fp32 (TDM/async staged)
    __shared__ unsigned short Bs[64 * 32];   // [n][k]  bf16 (transposed)

    const int tid      = threadIdx.x;
    const int wid      = tid >> 5;
    const int wm       = wid >> 1;           // 0..3 : 16-row sub-tile
    const int wn       = wid & 1;            // 0..1 : 32-col half
    const int lane     = tid & 31;
    const int half     = lane >> 4;
    const int l16      = lane & 15;
    const int blockRow = blockIdx.x * 64;
    const int colBase  = blockIdx.y * 64;

    // per-thread async A staging coords: 2 x 16B (rows r0 and r0+32)
    const int r0 = tid >> 3;                 // 0..31
    const int c0 = (tid & 7) * 4;            // 0,4,...,28
    const unsigned ldsA  = (unsigned)(uintptr_t)As;
    const unsigned ldsA0 = ldsA + (unsigned)((r0 * 32 + c0) * 4);
    const unsigned ldsA1 = ldsA0 + 32 * 32 * 4;
    const int row0g = blockRow + r0;
    const int row1g = row0g + 32;

    f32x8 acc0 = {};
    f32x8 acc1 = {};

    for (int k0 = 0; k0 < K; k0 += 32) {
        if (use_tdm) {
            // ---- one Tensor Data Mover op per block: 64x32 fp32 tile -> LDS
            if (wid == 0) {
                unsigned long long ga = (unsigned long long)(uintptr_t)A +
                    ((unsigned long long)blockRow * lda + acol + k0) * 4ull;
                unsigned dim0 = (unsigned)lda;              // tensor x-extent
                unsigned dim1 = (unsigned)(M - blockRow);   // rows left (OOB rows -> 0)
                unsigned long long st0 = (unsigned long long)lda;
                u32x4 g0;
                g0[0] = 1u;                                       // count=1 valid D#
                g0[1] = ldsA;                                     // lds_addr
                g0[2] = (unsigned)(ga & 0xFFFFFFFFull);           // global_addr lo
                g0[3] = (unsigned)((ga >> 32) & 0x1FFFFFFull) | (2u << 30); // hi + type=2
                i32x8 g1;
                g1[0] = (int)(2u << 16);                          // data_size=4B
                g1[1] = (int)((dim0 & 0xFFFFu) << 16);            // tensor_dim0 lo16
                g1[2] = (int)(((dim0 >> 16) & 0xFFFFu) | ((dim1 & 0xFFFFu) << 16));
                g1[3] = (int)(((dim1 >> 16) & 0xFFFFu) | (32u << 16)); // tile_dim0=32
                g1[4] = (int)64;                                  // tile_dim1=64
                g1[5] = (int)(unsigned)(st0 & 0xFFFFFFFFull);     // dim0_stride lo
                g1[6] = (int)(unsigned)((st0 >> 32) & 0xFFFFull); // dim0_stride hi
                g1[7] = 0;
                i32x4 z4 = {0, 0, 0, 0};
                i32x8 z8 = {0, 0, 0, 0, 0, 0, 0, 0};
                __builtin_amdgcn_tensor_load_to_lds(g0, g1, z4, z4, z8, 0);
                __builtin_amdgcn_s_wait_tensorcnt(0);
            }
        } else {
            // ---- per-lane async copies (ASYNCcnt), cache -> LDS directly
            if (row0g < M) {
                unsigned go = (unsigned)(((long)row0g * lda + acol + k0 + c0) * 4);
                asm volatile("global_load_async_to_lds_b128 %0, %1, %2"
                             :: "v"(ldsA0), "v"(go), "s"(A) : "memory");
            }
            if (row1g < M) {
                unsigned go = (unsigned)(((long)row1g * lda + acol + k0 + c0) * 4);
                asm volatile("global_load_async_to_lds_b128 %0, %1, %2"
                             :: "v"(ldsA1), "v"(go), "s"(A) : "memory");
            }
            asm volatile("s_wait_asynccnt 0x0" ::: "memory");
        }
        // ---- stage W tile 32x64, transposed + bf16
        #pragma unroll
        for (int i = 0; i < 8; ++i) {
            int e  = tid + i * 256;          // 0..2047
            int kk = e >> 6, n = e & 63;
            int gcn = colBase + n;
            float v = (gcn < N) ? W[(size_t)(k0 + kk) * ldw + gcn] : 0.f;
            Bs[n * 32 + kk] = f2bf(v);
        }
        __syncthreads();

        // ---- fragments per CDNA5 16-bit A(16x32)/B(32x16) VGPR layouts
        FragA fa, fb0, fb1;
        int arow = wm * 16 + l16;
        #pragma unroll
        for (int j = 0; j < 8; ++j) {
            int kk = ((j & 3) << 1) + (half << 3) + ((j >> 2) << 4);
            float x0 = As[arow * 32 + kk];
            float x1 = As[arow * 32 + kk + 1];
            if (gelu_in) { x0 = gelu_exact(x0); x1 = gelu_exact(x1); }
            fa.u[j] = (unsigned)f2bf(x0) | ((unsigned)f2bf(x1) << 16);
        }
        int bcol0 = wn * 32 + l16;
        int bcol1 = bcol0 + 16;
        #pragma unroll
        for (int j = 0; j < 8; ++j) {
            int kk = ((j & 3) << 1) + (half << 3) + ((j >> 2) << 4);
            fb0.u[j] = *(const unsigned int*)&Bs[bcol0 * 32 + kk];
            fb1.u[j] = *(const unsigned int*)&Bs[bcol1 * 32 + kk];
        }
        acc0 = __builtin_amdgcn_wmma_f32_16x16x32_bf16(
                   false, fa.v, false, fb0.v, (short)0, acc0, false, false);
        acc1 = __builtin_amdgcn_wmma_f32_16x16x32_bf16(
                   false, fa.v, false, fb1.v, (short)0, acc1, false, false);
        __syncthreads();
    }

    // ---- writeback: C/D layout -> VGPR r: M = r + half*8, N = lane%16
    int row0 = blockRow + wm * 16 + half * 8;
    #pragma unroll
    for (int s = 0; s < 2; ++s) {
        int gc = colBase + wn * 32 + s * 16 + l16;
        if (gc >= N) continue;
        float bv = bias ? bias[gc] : 0.f;
        const f32x8& acc = s ? acc1 : acc0;
        #pragma unroll
        for (int r = 0; r < 8; ++r) {
            int row = row0 + r;
            if (row < M) {
                float v = acc[r] + bv;
                if (gelu_out) v = gelu_exact(v);
                C[(size_t)row * ldc + ccol + gc] = v;
            }
        }
    }
}

// ---------------------------------------------------------------------------
// fill workspace region with a u32 value
// ---------------------------------------------------------------------------
__global__ void fill_u32_k(unsigned int* __restrict__ p, unsigned int v, int n) {
    int i = blockIdx.x * blockDim.x + threadIdx.x;
    if (i < n) p[i] = v;
}

// ---------------------------------------------------------------------------
// attention scores, one wave32 per edge (coalesced float4 gathers):
//   a[e,h] = (q[dst,h] . ke[src,h]) * prior[h] / sqrt(D)
// 8-lane head groups reduce via shfl_xor; leaders do sortable-uint atomicMax.
// ---------------------------------------------------------------------------
__global__ __launch_bounds__(256)
void score_k(const int* __restrict__ edges, int E,
             const float* __restrict__ kqv_dst,   // dst-type kqv, q at col 128
             const float* __restrict__ ke_src,    // [Nsrc,128]
             const float* __restrict__ prior,     // [4]
             int dstOff,
             float* __restrict__ attn,            // [E,4]
             unsigned int* __restrict__ segmax) { // [Ntot,4]
    int wid  = threadIdx.x >> 5;
    int lane = threadIdx.x & 31;
    int e = blockIdx.x * 8 + wid;
    if (e >= E) return;
    int src = edges[e], dst = edges[E + e];
    const float4* q  = (const float4*)(kqv_dst + (size_t)dst * 384 + 128);
    const float4* kk = (const float4*)(ke_src  + (size_t)src * 128);
    float4 qv = q[lane];
    float4 kv = kk[lane];
    float s = qv.x * kv.x + qv.y * kv.y + qv.z * kv.z + qv.w * kv.w;
    s += __shfl_xor(s, 1, 32);
    s += __shfl_xor(s, 2, 32);
    s += __shfl_xor(s, 4, 32);
    if ((lane & 7) == 0) {
        int h = lane >> 3;
        s *= prior[h] * 0.17677669529663687f;    // 1/sqrt(32)
        attn[(size_t)e * 4 + h] = s;
        atomicMax(&segmax[(size_t)(dst + dstOff) * 4 + h], fkey(s));
    }
}

// ---------------------------------------------------------------------------
// w = exp(a - max[gid]); den[gid] += w
// ---------------------------------------------------------------------------
__global__ void expden_k(const int* __restrict__ edges, int E, int dstOff,
                         float* __restrict__ attn,
                         const unsigned int* __restrict__ segmax,
                         float* __restrict__ den) {
    int i = blockIdx.x * blockDim.x + threadIdx.x;
    if (i >= E * 4) return;
    int e = i >> 2, h = i & 3;
    int gid = edges[E + e] + dstOff;
    float m = funkey(segmax[(size_t)gid * 4 + h]);
    float w = __expf(attn[(size_t)e * 4 + h] - m);
    attn[(size_t)e * 4 + h] = w;
    atomicAdd(&den[(size_t)gid * 4 + h], w);
}

// ---------------------------------------------------------------------------
// agg[gid, c] += ve[src, c] * w[e,h]/den[gid,h]   (one thread per edge-channel)
// ---------------------------------------------------------------------------
__global__ void agg_k(const int* __restrict__ edges, int E, int dstOff,
                      const float* __restrict__ attn,
                      const float* __restrict__ den,
                      const float* __restrict__ ve,
                      float* __restrict__ agg) {
    int i = blockIdx.x * blockDim.x + threadIdx.x;
    if (i >= E * 128) return;
    int e = i >> 7, c = i & 127, h = c >> 5;
    int src = edges[e];
    int gid = edges[E + e] + dstOff;
    float coef = attn[(size_t)e * 4 + h] / den[(size_t)gid * 4 + h];
    atomicAdd(&agg[(size_t)gid * 128 + c], ve[(size_t)src * 128 + c] * coef);
}

// ---------------------------------------------------------------------------
// out = LayerNorm( sigmoid(skip)*o + (1-sigmoid(skip))*x ), one wave32 per row
// ---------------------------------------------------------------------------
__global__ __launch_bounds__(128)
void skip_ln_k(const float* __restrict__ o, const float* __restrict__ xin,
               const float* __restrict__ skip, const float* __restrict__ g,
               const float* __restrict__ bb, float* __restrict__ out, int N) {
    int wid  = threadIdx.x >> 5;
    int lane = threadIdx.x & 31;
    int row  = blockIdx.x * 4 + wid;
    if (row >= N) return;
    float sk = 1.f / (1.f + __expf(-skip[0]));
    float y[4];
    float s = 0.f;
    #pragma unroll
    for (int j = 0; j < 4; ++j) {
        int c = lane * 4 + j;
        float v = sk * o[(size_t)row * 128 + c] + (1.f - sk) * xin[(size_t)row * 128 + c];
        y[j] = v; s += v;
    }
    for (int off = 16; off; off >>= 1) s += __shfl_xor(s, off, 32);
    float mu = s * (1.f / 128.f);
    float vs = 0.f;
    #pragma unroll
    for (int j = 0; j < 4; ++j) { float d = y[j] - mu; vs += d * d; }
    for (int off = 16; off; off >>= 1) vs += __shfl_xor(vs, off, 32);
    float inv = rsqrtf(vs * (1.f / 128.f) + 1e-5f);
    #pragma unroll
    for (int j = 0; j < 4; ++j) {
        int c = lane * 4 + j;
        out[(size_t)row * 128 + c] = (y[j] - mu) * inv * g[c] + bb[c];
    }
}

// ---------------------------------------------------------------------------
// host orchestration
// ---------------------------------------------------------------------------
static inline void launch_gemm(hipStream_t s,
                               const float* A, int lda, int acol,
                               const float* W, int ldw, const float* bias,
                               float* C, int ldc, int ccol,
                               int M, int K, int N, int gin, int gout, int tdm) {
    dim3 grid((M + 63) / 64, (N + 63) / 64);
    wmma_gemm_k<<<grid, 256, 0, s>>>(A, lda, acol, W, ldw, bias, C, ldc, ccol,
                                     M, N, K, gin, gout, tdm);
}

extern "C" void kernel_launch(void* const* d_in, const int* in_sizes, int n_in,
                              void* d_out, int out_size, void* d_ws, size_t ws_size,
                              hipStream_t stream) {
    int p = 0;
    const float* x_a = (const float*)d_in[p++];
    const float* x_b = (const float*)d_in[p++];
    const int*   e_ab = (const int*)d_in[p++];
    const int*   e_ba = (const int*)d_in[p++];

    const int NA = in_sizes[0] / 64;
    const int NB = in_sizes[1] / 64;
    const int EA = in_sizes[2] / 2;
    const int EB = in_sizes[3] / 2;
    const int NT = NA + NB;

    const float *inW[2], *inb[2];
    for (int t = 0; t < 2; ++t) { inW[t] = (const float*)d_in[p++]; inb[t] = (const float*)d_in[p++]; }

    struct Layer {
        const float *kqvW[2], *kqvb[2], *outW[2], *outb[2], *skip[2];
        const float *A[2], *Mm[2], *pr[2], *lng, *lnb;
    } Lp[2];
    for (int l = 0; l < 2; ++l) {
        for (int t = 0; t < 2; ++t) { Lp[l].kqvW[t] = (const float*)d_in[p++]; Lp[l].kqvb[t] = (const float*)d_in[p++]; }
        for (int t = 0; t < 2; ++t) { Lp[l].outW[t] = (const float*)d_in[p++]; Lp[l].outb[t] = (const float*)d_in[p++]; }
        for (int t = 0; t < 2; ++t) { Lp[l].skip[t] = (const float*)d_in[p++]; }
        for (int e = 0; e < 2; ++e) { Lp[l].A[e]  = (const float*)d_in[p++]; }
        for (int e = 0; e < 2; ++e) { Lp[l].Mm[e] = (const float*)d_in[p++]; }
        for (int e = 0; e < 2; ++e) { Lp[l].pr[e] = (const float*)d_in[p++]; }
        Lp[l].lng = (const float*)d_in[p++];
        Lp[l].lnb = (const float*)d_in[p++];
    }

    // workspace layout
    float* ws = (float*)d_ws;
    size_t o = 0;
    float* feat_a  = ws + o; o += (size_t)NA * 128;
    float* feat_b  = ws + o; o += (size_t)NB * 128;
    float* feat2_a = ws + o; o += (size_t)NA * 128;
    float* feat2_b = ws + o; o += (size_t)NB * 128;
    float* kqv_a   = ws + o; o += (size_t)NA * 384;
    float* kqv_b   = ws + o; o += (size_t)NB * 384;
    float* ke_ab   = ws + o; o += (size_t)NA * 128;   // reused as o_a after agg
    float* ve_ab   = ws + o; o += (size_t)NA * 128;
    float* ke_ba   = ws + o; o += (size_t)NB * 128;   // reused as o_b after agg
    float* ve_ba   = ws + o; o += (size_t)NB * 128;
    float* attn_ab = ws + o; o += (size_t)EA * 4;
    float* attn_ba = ws + o; o += (size_t)EB * 4;
    // contiguous zero-init block: segmax | den | agg
    unsigned int* segmax = (unsigned int*)(ws + o); o += (size_t)NT * 4;
    float* den = ws + o; o += (size_t)NT * 4;
    float* agg = ws + o; o += (size_t)NT * 128;

    float* out_a = (float*)d_out;
    float* out_b = (float*)d_out + (size_t)NA * 128;

    // ---- input projection + GELU (TDM-staged A tiles) ----
    launch_gemm(stream, x_a, 64, 0, inW[0], 128, inb[0], feat_a, 128, 0, NA, 64, 128, 0, 1, 1);
    launch_gemm(stream, x_b, 64, 0, inW[1], 128, inb[1], feat_b, 128, 0, NB, 64, 128, 0, 1, 1);

    float* cur_a = feat_a;  float* cur_b = feat_b;
    float* nxt_a = feat2_a; float* nxt_b = feat2_b;

    for (int l = 0; l < 2; ++l) {
        const Layer& L = Lp[l];

        // kqv projections (TDM)
        launch_gemm(stream, cur_a, 128, 0, L.kqvW[0], 384, L.kqvb[0], kqv_a, 384, 0, NA, 128, 384, 0, 0, 1);
        launch_gemm(stream, cur_b, 128, 0, L.kqvW[1], 384, L.kqvb[1], kqv_b, 384, 0, NB, 128, 384, 0, 0, 1);

        // per-head relation transforms: ke = k @ A[h], ve = v @ M[h]  (async path)
        for (int h = 0; h < 4; ++h) {
            launch_gemm(stream, kqv_a, 384, h * 32,       L.A[0]  + h * 1024, 32, nullptr, ke_ab, 128, h * 32, NA, 32, 32, 0, 0, 0);
            launch_gemm(stream, kqv_a, 384, 256 + h * 32, L.Mm[0] + h * 1024, 32, nullptr, ve_ab, 128, h * 32, NA, 32, 32, 0, 0, 0);
            launch_gemm(stream, kqv_b, 384, h * 32,       L.A[1]  + h * 1024, 32, nullptr, ke_ba, 128, h * 32, NB, 32, 32, 0, 0, 0);
            launch_gemm(stream, kqv_b, 384, 256 + h * 32, L.Mm[1] + h * 1024, 32, nullptr, ve_ba, 128, h * 32, NB, 32, 32, 0, 0, 0);
        }

        // zero segmax/den/agg (contiguous)
        {
            int n = NT * 4 + NT * 4 + NT * 128;
            fill_u32_k<<<(n + 255) / 256, 256, 0, stream>>>((unsigned int*)segmax, 0u, n);
        }

        // scores + segment max    (a->b: dst type 'b', offset NA; b->a: dst 'a', offset 0)
        score_k<<<(EA + 7) / 8, 256, 0, stream>>>(e_ab, EA, kqv_b, ke_ab, L.pr[0], NA, attn_ab, segmax);
        score_k<<<(EB + 7) / 8, 256, 0, stream>>>(e_ba, EB, kqv_a, ke_ba, L.pr[1], 0,  attn_ba, segmax);

        // exp + denominators
        expden_k<<<(EA * 4 + 255) / 256, 256, 0, stream>>>(e_ab, EA, NA, attn_ab, segmax, den);
        expden_k<<<(EB * 4 + 255) / 256, 256, 0, stream>>>(e_ba, EB, 0,  attn_ba, segmax, den);

        // weighted aggregate
        agg_k<<<((size_t)EA * 128 + 255) / 256, 256, 0, stream>>>(e_ab, EA, NA, attn_ab, den, ve_ab, agg);
        agg_k<<<((size_t)EB * 128 + 255) / 256, 256, 0, stream>>>(e_ba, EB, 0,  attn_ba, den, ve_ba, agg);

        // out projection with input-GELU (reuse ke buffers as o; async path)
        float* o_a = ke_ab;
        float* o_b = ke_ba;
        launch_gemm(stream, agg,                    128, 0, L.outW[0], 128, L.outb[0], o_a, 128, 0, NA, 128, 128, 1, 0, 0);
        launch_gemm(stream, agg + (size_t)NA * 128, 128, 0, L.outW[1], 128, L.outb[1], o_b, 128, 0, NB, 128, 128, 1, 0, 0);

        // skip-gate + LayerNorm
        float* dst_a = (l == 1) ? out_a : nxt_a;
        float* dst_b = (l == 1) ? out_b : nxt_b;
        skip_ln_k<<<(NA + 3) / 4, 128, 0, stream>>>(o_a, cur_a, L.skip[0], L.lng, L.lnb, dst_a, NA);
        skip_ln_k<<<(NB + 3) / 4, 128, 0, stream>>>(o_b, cur_b, L.skip[1], L.lng, L.lnb, dst_b, NB);

        float* ta = cur_a; cur_a = nxt_a; nxt_a = ta;
        float* tb = cur_b; cur_b = nxt_b; nxt_b = tb;
    }
}
